// MultiheadAttention_8881992368163
// MI455X (gfx1250) — compile-verified, matrix operand order checked by reading, not run on hardware
//
#include <hip/hip_runtime.h>

typedef __bf16 bf16;
typedef __attribute__((ext_vector_type(16))) bf16 v16bf;
typedef __attribute__((ext_vector_type(8)))  bf16 v8bf;
typedef __attribute__((ext_vector_type(8)))  float v8f;

constexpr int Bc = 4, Sc = 2048, Hc = 1024, NHc = 16, DHc = 64;
constexpr int Mc = Bc * Sc;  // 8192 rows (B*S)

// ---------------------------------------------------------------------------
// CDNA5 async global->LDS copy (ASYNCcnt path, CDNA5 ISA).
// lds_off: byte offset in LDS (low 32 bits of flat shared address per ISA
// 10.2 aperture rules). Each active lane moves 16B without touching VGPRs.
// ---------------------------------------------------------------------------
__device__ __forceinline__ void async_load_b128(uint32_t lds_off, const void* gaddr) {
  asm volatile("global_load_async_to_lds_b128 %0, %1, off"
               :: "v"(lds_off), "v"(gaddr) : "memory");
}
__device__ __forceinline__ void wait_async0() {
  asm volatile("s_wait_asynccnt 0x0" ::: "memory");
}

// ---------------------------------------------------------------------------
// bf16 16x32 A/B fragment loader (CDNA5 ISA layout): lanes 0-15 hold
// K=0..7 / K=16..23, lanes 16-31 hold K=8..15 / K=24..31; works for global
// (global_load_b128) and LDS (ds_load_b128) pointers after inlining.
// ---------------------------------------------------------------------------
__device__ __forceinline__ v16bf load_frag(const bf16* p, int ld, int lane) {
  const int r  = lane & 15;
  const int hk = (lane & 16) >> 1;         // 0 or 8
  const bf16* q = p + (size_t)r * ld + hk;
  v8bf lo = *(const v8bf*)(q);
  v8bf hi = *(const v8bf*)(q + 16);
  v16bf f;
#pragma unroll
  for (int i = 0; i < 8; ++i) { f[i] = lo[i]; f[i + 8] = hi[i]; }
  return f;
}

// ---------------------------------------------------------------------------
// f32 -> bf16 convert
// ---------------------------------------------------------------------------
__global__ __launch_bounds__(256) void cvt_f32_bf16(const float* __restrict__ src,
                                                    bf16* __restrict__ dst, int n) {
  int i = (blockIdx.x * blockDim.x + threadIdx.x) * 4;
  if (i < n) {
    float4 v = *(const float4*)(src + i);
    dst[i + 0] = (bf16)v.x; dst[i + 1] = (bf16)v.y;
    dst[i + 2] = (bf16)v.z; dst[i + 3] = (bf16)v.w;
  }
}

// ---------------------------------------------------------------------------
// GEMM: out[m][n] = sum_k A[m][k] * W[n][k] + bias[n]   (M=8192, N=K=1024)
// Block tile 128x128 (8 waves, each 32x64 = 2x4 WMMA accums), K-panels
// double-buffered in LDS via async loads => ~64 FLOP per L2 byte.
// mode 0: bf16 *scale (Q)   mode 1: bf16 (K)   mode 2: bf16 Vt[b,h,d,s]
// mode 3: f32 row-major (final output)
// ---------------------------------------------------------------------------
constexpr int BM = 128, BN = 128, BK = 32;
constexpr int LDP = 40;   // padded LDS row stride (elems): 80B, bank-skewed

__global__ __launch_bounds__(256) void gemm_bf16(
    const bf16* __restrict__ A, const bf16* __restrict__ W,
    const float* __restrict__ bias, bf16* __restrict__ outb,
    float* __restrict__ outf, int mode, float scale) {
  __shared__ __align__(16) bf16 sA[2][BM * LDP];
  __shared__ __align__(16) bf16 sB[2][BN * LDP];

  const int tid  = threadIdx.x;
  const int lane = tid & 31;
  const int w    = tid >> 5;
  const int wm   = (w & 3) * 32;   // wave tile: 32 rows x 64 cols
  const int wn   = (w >> 2) * 64;
  const int m_base = blockIdx.x * BM;
  const int n_base = blockIdx.y * BN;

  auto issue = [&](int buf, int k0) {
#pragma unroll
    for (int i = 0; i < 2; ++i) {
      const int ci  = tid + i * 256;        // 0..511 : 128 rows x 4 16B-chunks
      const int row = ci >> 2, part = ci & 3;
      async_load_b128((uint32_t)(uintptr_t)&sA[buf][row * LDP + part * 8],
                      A + (size_t)(m_base + row) * Hc + k0 + part * 8);
      async_load_b128((uint32_t)(uintptr_t)&sB[buf][row * LDP + part * 8],
                      W + (size_t)(n_base + row) * Hc + k0 + part * 8);
    }
  };

  v8f acc[2][4] = {};
  issue(0, 0);
  wait_async0();
  __syncthreads();

  constexpr int NK = Hc / BK;   // 32
  for (int kt = 0; kt < NK; ++kt) {
    const int cur = kt & 1;
    if (kt + 1 < NK) issue(cur ^ 1, (kt + 1) * BK);

    v16bf af[2], bfr[4];
#pragma unroll
    for (int i = 0; i < 2; ++i)
      af[i] = load_frag(&sA[cur][(wm + i * 16) * LDP], LDP, lane);
#pragma unroll
    for (int j = 0; j < 4; ++j)
      bfr[j] = load_frag(&sB[cur][(wn + j * 16) * LDP], LDP, lane);
#pragma unroll
    for (int i = 0; i < 2; ++i)
#pragma unroll
      for (int j = 0; j < 4; ++j)
        acc[i][j] = __builtin_amdgcn_wmma_f32_16x16x32_bf16(
            false, af[i], false, bfr[j], (short)0, acc[i][j], false, false);

    if (kt + 1 < NK) wait_async0();
    __syncthreads();
  }

  const int n  = lane & 15;
  const int mo = (lane & 16) ? 8 : 0;
#pragma unroll
  for (int i = 0; i < 2; ++i) {
#pragma unroll
    for (int j = 0; j < 4; ++j) {
      const int gn = n_base + wn + j * 16 + n;
      const float bv = bias[gn];
#pragma unroll
      for (int r = 0; r < 8; ++r) {
        const int gm = m_base + wm + i * 16 + mo + r;
        float v = acc[i][j][r] + bv;
        if (mode == 0) {
          outb[(size_t)gm * Hc + gn] = (bf16)(v * scale);
        } else if (mode == 1) {
          outb[(size_t)gm * Hc + gn] = (bf16)v;
        } else if (mode == 2) {
          const int bb = gm >> 11, ss = gm & (Sc - 1);
          const int hh = gn >> 6,  dd = gn & (DHc - 1);
          outb[((size_t)(bb * NHc + hh) * DHc + dd) * Sc + ss] = (bf16)v;
        } else {
          outf[(size_t)gm * Hc + gn] = v;
        }
      }
    }
  }
}

// ---------------------------------------------------------------------------
// Flash attention: 8 waves/block = 128 q-rows of one (b,h); all waves share
// the same 64-wide K/V chunk, staged in LDS (double-buffered async loads) so
// K/V L2 traffic drops 8x vs per-wave loads. Online softmax; scores never hit
// HBM (a materialized score tensor would cost ~2GB of HBM round trip ~90us).
// ---------------------------------------------------------------------------
constexpr int LKV = 72;   // padded LDS row stride (elems): 144B, bank-skewed

__global__ __launch_bounds__(256) void flash_attn(
    const bf16* __restrict__ Q, const bf16* __restrict__ Kg,
    const bf16* __restrict__ Vt, const int* __restrict__ amask,
    const int* __restrict__ kmask, bf16* __restrict__ AO) {
  __shared__ __align__(16) bf16 sK[2][64 * LKV];   // K chunk: 64 keys x 64 dh
  __shared__ __align__(16) bf16 sV[2][64 * LKV];   // V chunk: 64 dh x 64 keys
  __shared__ __align__(16) bf16 sP[8][16 * LKV];   // per-wave P staging

  const int tid  = threadIdx.x;
  const int lane = tid & 31;
  const int w    = tid >> 5;
  const int b    = blockIdx.z;
  const int h    = blockIdx.y;
  const int q0   = blockIdx.x * 128 + w * 16;
  const int n    = lane & 15;
  const int mo   = (lane & 16) ? 8 : 0;

  const bf16* Kbase = Kg + ((size_t)(b * Sc)) * Hc + h * DHc;
  const bf16* Vbase = Vt + (size_t)((b * NHc + h) * DHc) * Sc;

  auto issueKV = [&](int buf, int kc) {
#pragma unroll
    for (int i = 0; i < 2; ++i) {
      const int ci  = tid + i * 256;      // 0..511 : 64 rows x 8 16B-chunks
      const int row = ci >> 3, part = ci & 7;
      async_load_b128((uint32_t)(uintptr_t)&sK[buf][row * LKV + part * 8],
                      Kbase + (size_t)(kc + row) * Hc + part * 8);
      async_load_b128((uint32_t)(uintptr_t)&sV[buf][row * LKV + part * 8],
                      Vbase + (size_t)row * Sc + kc + part * 8);
    }
  };

  const bf16* Qp = Q + ((size_t)(b * Sc + q0)) * Hc + h * DHc;
  v16bf qf[2];
#pragma unroll
  for (int t = 0; t < 2; ++t) qf[t] = load_frag(Qp + t * 32, Hc, lane);

  float mrow[8], lsum[8];
#pragma unroll
  for (int r = 0; r < 8; ++r) { mrow[r] = -3.0e38f; lsum[r] = 0.f; }
  v8f o[4] = {};

  bf16* P = &sP[w][0];

  issueKV(0, 0);
  wait_async0();
  __syncthreads();

  for (int kc = 0; kc < Sc; kc += 64) {
    const int cur = (kc >> 6) & 1;
    if (kc + 64 < Sc) issueKV(cur ^ 1, kc + 64);

    // ---- scores: S = Q (16x64) * K_chunk^T (64x64)
    v8f sc[4];
#pragma unroll
    for (int j = 0; j < 4; ++j) {
      v8f s = {};
#pragma unroll
      for (int t = 0; t < 2; ++t)
        s = __builtin_amdgcn_wmma_f32_16x16x32_bf16(
            false, qf[t], false,
            load_frag(&sK[cur][(j * 16) * LKV + t * 32], LKV, lane),
            (short)0, s, false, false);
      sc[j] = s;
    }
    // ---- masks: where(mask) -> -1e12 (reference semantics)
#pragma unroll
    for (int j = 0; j < 4; ++j) {
      const int k  = kc + j * 16 + n;
      const int km = kmask[b * Sc + k];
#pragma unroll
      for (int r = 0; r < 8; ++r) {
        const int am = amask[(size_t)(q0 + mo + r) * Sc + k];
        if (am | km) sc[j][r] = -1.0e12f;
      }
    }
    // ---- online softmax (row = 16 lanes of a half; xor 1/2/4/8 stays in-half)
#pragma unroll
    for (int r = 0; r < 8; ++r) {
      float v = fmaxf(fmaxf(sc[0][r], sc[1][r]), fmaxf(sc[2][r], sc[3][r]));
      v = fmaxf(v, __shfl_xor(v, 1));
      v = fmaxf(v, __shfl_xor(v, 2));
      v = fmaxf(v, __shfl_xor(v, 4));
      v = fmaxf(v, __shfl_xor(v, 8));
      const float mnew = fmaxf(mrow[r], v);
      const float corr = __expf(mrow[r] - mnew);
      float ps = 0.f;
#pragma unroll
      for (int j = 0; j < 4; ++j) { float p = __expf(sc[j][r] - mnew); sc[j][r] = p; ps += p; }
      ps += __shfl_xor(ps, 1);
      ps += __shfl_xor(ps, 2);
      ps += __shfl_xor(ps, 4);
      ps += __shfl_xor(ps, 8);
      lsum[r] = lsum[r] * corr + ps;
      mrow[r] = mnew;
#pragma unroll
      for (int j = 0; j < 4; ++j) o[j][r] *= corr;
    }
    // ---- P (C-layout f32) -> LDS bf16, reload as A-fragments
#pragma unroll
    for (int j = 0; j < 4; ++j)
#pragma unroll
      for (int r = 0; r < 8; ++r)
        P[(mo + r) * LKV + j * 16 + n] = (bf16)sc[j][r];
    asm volatile("s_wait_dscnt 0x0" ::: "memory");
    // ---- O += P (16x64) * V_chunk (64x64) from LDS
#pragma unroll
    for (int t = 0; t < 2; ++t) {
      v16bf pf = load_frag(P + t * 32, LKV, lane);
#pragma unroll
      for (int j = 0; j < 4; ++j)
        o[j] = __builtin_amdgcn_wmma_f32_16x16x32_bf16(
            false, pf, false,
            load_frag(&sV[cur][(j * 16) * LKV + t * 32], LKV, lane),
            (short)0, o[j], false, false);
    }

    if (kc + 64 < Sc) wait_async0();
    __syncthreads();
  }

  // ---- normalize and store attention output (bf16, row-major [B*S, H])
#pragma unroll
  for (int r = 0; r < 8; ++r) {
    const float inv = 1.0f / lsum[r];
#pragma unroll
    for (int j = 0; j < 4; ++j)
      AO[((size_t)(b * Sc + q0 + mo + r)) * Hc + h * DHc + j * 16 + n] =
          (bf16)(o[j][r] * inv);
  }
}

// ---------------------------------------------------------------------------
// Launcher
// ---------------------------------------------------------------------------
extern "C" void kernel_launch(void* const* d_in, const int* in_sizes, int n_in,
                              void* d_out, int out_size, void* d_ws, size_t ws_size,
                              hipStream_t stream) {
  const float* query  = (const float*)d_in[0];
  const float* keyval = (const float*)d_in[1];
  const int*   kign   = (const int*)  d_in[2];
  const int*   amask  = (const int*)  d_in[3];
  const float* Wq = (const float*)d_in[4];
  const float* bq = (const float*)d_in[5];
  const float* Wk = (const float*)d_in[6];
  const float* bk = (const float*)d_in[7];
  const float* Wv = (const float*)d_in[8];
  const float* bv = (const float*)d_in[9];
  const float* Wo = (const float*)d_in[10];
  const float* bo = (const float*)d_in[11];
  float* out = (float*)d_out;

  bf16* ws = (bf16*)d_ws;
  const size_t E  = (size_t)Mc * Hc;   // 8,388,608 elems
  const size_t WE = (size_t)Hc * Hc;   // 1,048,576 elems
  bf16* Xq  = ws;          // query bf16
  bf16* Xkv = Xq  + E;     // key_value bf16
  bf16* Wqb = Xkv + E;
  bf16* Wkb = Wqb + WE;
  bf16* Wvb = Wkb + WE;
  bf16* Wob = Wvb + WE;
  bf16* Qb  = Wob + WE;    // scaled Q, bf16 [B*S, H]
  bf16* Kb  = Qb  + E;     // K, bf16 [B*S, H]
  bf16* Vt  = Kb  + E;     // V transposed, bf16 [B, NH, DH, S]
  bf16* AO  = Vt  + E;     // attention output, bf16 [B*S, H]

  // converts
  cvt_f32_bf16<<<dim3((unsigned)(E  / 1024)), dim3(256), 0, stream>>>(query,  Xq,  (int)E);
  cvt_f32_bf16<<<dim3((unsigned)(E  / 1024)), dim3(256), 0, stream>>>(keyval, Xkv, (int)E);
  cvt_f32_bf16<<<dim3((unsigned)(WE / 1024)), dim3(256), 0, stream>>>(Wq, Wqb, (int)WE);
  cvt_f32_bf16<<<dim3((unsigned)(WE / 1024)), dim3(256), 0, stream>>>(Wk, Wkb, (int)WE);
  cvt_f32_bf16<<<dim3((unsigned)(WE / 1024)), dim3(256), 0, stream>>>(Wv, Wvb, (int)WE);
  cvt_f32_bf16<<<dim3((unsigned)(WE / 1024)), dim3(256), 0, stream>>>(Wo, Wob, (int)WE);

  // projections (block tile 128x128)
  dim3 gg(Mc / BM, Hc / BN), gb(256);
  gemm_bf16<<<gg, gb, 0, stream>>>(Xq,  Wqb, bq, Qb, nullptr, 0, 0.125f);  // Q, /sqrt(DH)
  gemm_bf16<<<gg, gb, 0, stream>>>(Xkv, Wkb, bk, Kb, nullptr, 1, 1.0f);    // K
  gemm_bf16<<<gg, gb, 0, stream>>>(Xkv, Wvb, bv, Vt, nullptr, 2, 1.0f);    // V -> Vt

  // fused attention (128 q-rows per block)
  flash_attn<<<dim3(Sc / 128, NHc, Bc), dim3(256), 0, stream>>>(Qb, Kb, Vt, amask, kign, AO);

  // output projection -> f32
  gemm_bf16<<<gg, gb, 0, stream>>>(AO, Wob, bo, nullptr, out, 3, 1.0f);
}
